// BertLinearSelfAttention_17446157156418
// MI455X (gfx1250) — compile-verified, matrix operand order checked by reading, not run on hardware
//
#include <hip/hip_runtime.h>
#include <hip/hip_bf16.h>
#include <stdint.h>

// ---------------------------------------------------------------------------
// BERT linear self-attention (no softmax: probs = (Q K^T) * binary key mask)
// B=2, S=2048, D=1024, H=16, HD=64.
//
// Stage 0: f32->bf16 converts + mask binarize.
// Stage 1: QKV projections via v_wmma_f32_16x16x32_bf16, register
//          double-buffered fragment loads (overlap VMEM with WMMA).
// Stage 2: flash-style attention; K / V^T tiles staged into double-buffered
//          LDS by the Tensor Data Mover (tensor_load_to_lds + s_wait_tensorcnt)
//          issued by wave 0; P tiles transposed through per-wave LDS.
// ---------------------------------------------------------------------------

typedef __attribute__((ext_vector_type(16))) __bf16       bf16x16;
typedef __attribute__((ext_vector_type(8)))  float        f32x8;
typedef __attribute__((ext_vector_type(4)))  unsigned int u32x4;
typedef __attribute__((ext_vector_type(8)))  int          i32x8;
typedef __attribute__((ext_vector_type(4)))  int          i32x4;

#if defined(__gfx1250__) && __has_builtin(__builtin_amdgcn_tensor_load_to_lds)
#define USE_TDM 1
#else
#define USE_TDM 0
#endif

#define DEV __device__ __forceinline__

union FragA {
  bf16x16 v;
  uint4   q[2];
};

// Load a 16x32 bf16 A-style fragment from row-major data (ld elements/row).
// Per ISA dense 16-bit A layout: lanes<16 hold K {c0+0..7, c0+16..23},
// lanes>=16 hold K {c0+8..15, c0+24..31}; row = row0 + (lane&15).
DEV void load_fragA(FragA& f, const __bf16* base, int row0, int ld, int c0, int lane) {
  const int m    = lane & 15;
  const int half = (lane >> 4) & 1;
  const __bf16* p = base + (size_t)(row0 + m) * (size_t)ld + c0 + half * 8;
  f.q[0] = *(const uint4*)(p);
  f.q[1] = *(const uint4*)(p + 16);
}

DEV f32x8 wmma_bf16(const bf16x16 a, const bf16x16 b, f32x8 c) {
  return __builtin_amdgcn_wmma_f32_16x16x32_bf16(false, a, false, b, (short)0, c,
                                                 false, false);
}

DEV f32x8 zero8() {
  f32x8 z = {0.f, 0.f, 0.f, 0.f, 0.f, 0.f, 0.f, 0.f};
  return z;
}

DEV uint4 pack_bf16x8(f32x8 x) {
  union { uint4 q; __bf16 b[8]; } u;
#pragma unroll
  for (int i = 0; i < 8; ++i) u.b[i] = (__bf16)x[i];
  return u.q;
}

#if USE_TDM
// Issue a TDM 2-D tile load (bf16 elements) into LDS.
//   tensor: td0 (contiguous) x td1 rows, row stride = stride0 (elements)
//   tile:   t0 (contiguous)  x t1 rows, starting at gsrc
// D# layout per CDNA5 ISA 8.3/8.4 (group0: count/lds/global/type,
// group1: data_size, dims, tile dims, dim0 stride). Groups 2/3 zero (2-D).
// This toolchain exposes the 6-arg builtin: (g0, g1, g2, g3, g4, cpol).
DEV void tdm_load_tile_2d(unsigned lds_off, const __bf16* gsrc,
                          unsigned td0, unsigned td1, unsigned stride0,
                          unsigned t0, unsigned t1) {
  const unsigned long long ga = (unsigned long long)(uintptr_t)gsrc;
  u32x4 g0;
  g0[0] = 1u;                                   // count=1, user descriptor
  g0[1] = lds_off;                              // lds_addr (bytes)
  g0[2] = (unsigned)(ga & 0xffffffffu);         // global_addr[31:0]
  g0[3] = (unsigned)((ga >> 32) & 0x01ffffffu)  // global_addr[56:32]
          | (2u << 30);                         // type = 2 ("image")
  i32x8 g1;
  g1[0] = (int)(1u << 16);                      // data_size=1 -> 2 bytes
  g1[1] = (int)(td0 << 16);                     // tensor_dim0[15:0]
  g1[2] = (int)((td0 >> 16) | (td1 << 16));     // dim0[31:16] | dim1[15:0]
  g1[3] = (int)((td1 >> 16) | (t0 << 16));      // dim1[31:16] | tile_dim0
  g1[4] = (int)(t1 & 0xffffu);                  // tile_dim1 (tile_dim2=0)
  g1[5] = (int)stride0;                         // tensor_dim0_stride[31:0]
  g1[6] = 0;
  g1[7] = 0;
  i32x4 z4 = {0, 0, 0, 0};
  i32x8 z8 = {0, 0, 0, 0, 0, 0, 0, 0};
  __builtin_amdgcn_tensor_load_to_lds(g0, g1, z4, z4, z8, 0);
}
#endif

// ---------------------------------------------------------------------------
// Stage 0: elementwise converts
// ---------------------------------------------------------------------------
__global__ void cvt_f32_to_bf16(const float* __restrict__ in,
                                __bf16* __restrict__ out, int n4) {
  int i = blockIdx.x * blockDim.x + threadIdx.x;
  if (i >= n4) return;
  float4 f = ((const float4*)in)[i];
  union { uint2 u; __bf16 b[4]; } o;
  o.b[0] = (__bf16)f.x; o.b[1] = (__bf16)f.y;
  o.b[2] = (__bf16)f.z; o.b[3] = (__bf16)f.w;
  ((uint2*)out)[i] = o.u;
}

__global__ void binarize_mask(const float* __restrict__ in,
                              float* __restrict__ out, int n) {
  int i = blockIdx.x * blockDim.x + threadIdx.x;
  if (i < n) out[i] = (in[i] >= 0.f) ? 1.f : 0.f;
}

// ---------------------------------------------------------------------------
// Stage 1: QKV projection GEMM.  D = A(rows M x 1024) * Bm(rows N x 1024)^T.
// WG = 256 threads = 8 waves; WG tile 64(M) x 128(N); wave tile 16 x 64.
// Register double-buffered fragments: issue chunk k+1 loads, then run chunk
// k's WMMAs so waits are partial (VMEM hidden under matrix pipe).
// mode 0 (Q/K): A=W (M=o=1024), Bm=X (N=s=4096); store [B,H,s,hd] bf16.
// mode 1 (V):   A=X (M=s=4096), Bm=W (N=o=1024); store V^T [B,H,hd,s] bf16.
// ---------------------------------------------------------------------------
__global__ __launch_bounds__(256) void qkv_gemm(const __bf16* __restrict__ A,
                                                const __bf16* __restrict__ Bm,
                                                const float* __restrict__ bias,
                                                __bf16* __restrict__ out,
                                                int mode) {
  const int lane = threadIdx.x & 31;
  const int wave = threadIdx.x >> 5;                 // 0..7
  const int mrow = blockIdx.y * 64 + (wave & 3) * 16;
  const int ncol = blockIdx.x * 128 + (wave >> 2) * 64;
  const int half = (lane >> 4) & 1;
  const int m16  = lane & 15;

  f32x8 acc[4];
#pragma unroll
  for (int t = 0; t < 4; ++t) acc[t] = zero8();

  FragA a[2];
  FragA b[2][4];
  load_fragA(a[0], A, mrow, 1024, 0, lane);
#pragma unroll
  for (int t = 0; t < 4; ++t) load_fragA(b[0][t], Bm, ncol + t * 16, 1024, 0, lane);

  for (int kt = 0; kt < 1024; kt += 32) {
    const int cur = (kt >> 5) & 1;
    const int nxt = cur ^ 1;
    if (kt + 32 < 1024) {
      load_fragA(a[nxt], A, mrow, 1024, kt + 32, lane);
#pragma unroll
      for (int t = 0; t < 4; ++t)
        load_fragA(b[nxt][t], Bm, ncol + t * 16, 1024, kt + 32, lane);
    }
#pragma unroll
    for (int t = 0; t < 4; ++t)
      acc[t] = wmma_bf16(a[cur].v, b[cur][t].v, acc[t]);
  }

#pragma unroll
  for (int t = 0; t < 4; ++t) {
    if (mode == 0) {
      // C tile is (M=o, N=s): lane has fixed s, 8 consecutive o values.
      const int s_glob = ncol + t * 16 + m16;
      const int o_base = mrow + half * 8;
      float4 b0 = *(const float4*)(bias + o_base);
      float4 b1 = *(const float4*)(bias + o_base + 4);
      acc[t][0] += b0.x; acc[t][1] += b0.y; acc[t][2] += b0.z; acc[t][3] += b0.w;
      acc[t][4] += b1.x; acc[t][5] += b1.y; acc[t][6] += b1.z; acc[t][7] += b1.w;
      const int bi = s_glob >> 11, ss = s_glob & 2047;
      const int h  = o_base >> 6,  hd = o_base & 63;
      uint4 pk = pack_bf16x8(acc[t]);
      *(uint4*)(out + ((((size_t)bi * 16 + h) * 2048 + ss) * 64 + hd)) = pk;
    } else {
      // C tile is (M=s, N=o): lane has fixed o, 8 consecutive s values.
      const int o_glob = ncol + t * 16 + m16;
      const int s_base = mrow + half * 8;
      const float bb = bias[o_glob];
#pragma unroll
      for (int r = 0; r < 8; ++r) acc[t][r] += bb;
      const int bi = s_base >> 11, ss = s_base & 2047;
      const int h  = o_glob >> 6,  hd = o_glob & 63;
      uint4 pk = pack_bf16x8(acc[t]);
      *(uint4*)(out + ((((size_t)bi * 16 + h) * 64 + hd) * 2048 + ss)) = pk;
    }
  }
}

// ---------------------------------------------------------------------------
// Stage 2: attention.  grid.x = (B*H) * (S/64); WG = 128 threads = 4 waves;
// each wave owns one 16-query strip, accumulates ctx^T (64hd x 16q) over all
// 2048 keys in chunks of 32.  K / V^T chunk tiles staged into double-buffered
// LDS by the TDM (wave 0), all waves consume via ds_load_b128 fragments.
// ---------------------------------------------------------------------------
__global__ __launch_bounds__(128) void attn_kernel(const __bf16* __restrict__ Q,
                                                   const __bf16* __restrict__ K,
                                                   const __bf16* __restrict__ Vt,
                                                   const float* __restrict__ maskb,
                                                   float* __restrict__ out) {
  const int lane = threadIdx.x & 31;
  const int wave = threadIdx.x >> 5;     // 0..3
  const int tid  = threadIdx.x;
  const int qb   = blockIdx.x & 31;      // query block (S/64 = 32)
  const int bh   = blockIdx.x >> 5;      // 0..31
  const int b    = bh >> 4;
  const int h    = bh & 15;
  const int half = (lane >> 4) & 1;
  const int m16  = lane & 15;
  const int q0   = qb * 64 + wave * 16;

  const __bf16* Qh = Q  + (size_t)bh * 2048 * 64;
  const __bf16* Kh = K  + (size_t)bh * 2048 * 64;
  const __bf16* Vh = Vt + (size_t)bh * 64 * 2048;
  const float*  mrow = maskb + b * 2048;

  // Double-buffered K (32 keys x 64 hd) and V^T (64 hd x 32 keys) tiles.
  __shared__ __attribute__((aligned(16))) __bf16 Ks[2][32][64];
  __shared__ __attribute__((aligned(16))) __bf16 Vs[2][64][32];
  // Per-wave P transpose buffer: 16 q rows x 32 k cols, padded to 40.
  __shared__ __attribute__((aligned(16))) __bf16 Pl[4][16][40];

  // Stage one k-chunk (32 keys) into LDS buffer `buf`.
  auto issue_tiles = [&](int buf, int kt) {
#if USE_TDM
    if (wave == 0) {
      tdm_load_tile_2d((unsigned)(uintptr_t)&Ks[buf][0][0], Kh + (size_t)kt * 64,
                       /*td0=*/64, /*td1=*/2048, /*stride0=*/64,
                       /*t0=*/64, /*t1=*/32);
      tdm_load_tile_2d((unsigned)(uintptr_t)&Vs[buf][0][0], Vh + kt,
                       /*td0=*/2048, /*td1=*/64, /*stride0=*/2048,
                       /*t0=*/32, /*t1=*/64);
    }
#else
    // Fallback: cooperative copy. K tile is contiguous 4KB; V tile is 64
    // rows of 64B (stride 2048 elements).
    {
      const uint4* src = (const uint4*)(Kh + (size_t)kt * 64);
      uint4* dst = (uint4*)&Ks[buf][0][0];
      dst[tid * 2]     = src[tid * 2];
      dst[tid * 2 + 1] = src[tid * 2 + 1];
    }
    {
      const int row = tid >> 1, part = tid & 1;
      const uint4* src = (const uint4*)(Vh + (size_t)row * 2048 + kt + part * 16);
      uint4* dst = (uint4*)&Vs[buf][row][part * 16];
      dst[0] = src[0];
      dst[1] = src[1];
    }
#endif
  };
  auto finish_tiles = [&]() {
#if USE_TDM
    if (wave == 0) __builtin_amdgcn_s_wait_tensorcnt((short)0);
#endif
    __syncthreads();
  };

  FragA qf0, qf1;                         // B-frags of Q^T == A-frags of Q
  load_fragA(qf0, Qh, q0, 64, 0,  lane);
  load_fragA(qf1, Qh, q0, 64, 32, lane);

  f32x8 ctx[4];
#pragma unroll
  for (int t = 0; t < 4; ++t) ctx[t] = zero8();

  issue_tiles(0, 0);
  finish_tiles();

  for (int kt = 0; kt < 2048; kt += 32) {
    const int cur = (kt >> 5) & 1;
    if (kt + 32 < 2048) issue_tiles(cur ^ 1, kt + 32);

    // ---- S^T = K_tile @ Q^T, masked, packed to per-wave LDS P tile ----
#pragma unroll
    for (int sub = 0; sub < 2; ++sub) {
      f32x8 sacc = zero8();
      FragA kf;
      load_fragA(kf, &Ks[cur][0][0], sub * 16, 64, 0, lane);
      sacc = wmma_bf16(kf.v, qf0.v, sacc);
      load_fragA(kf, &Ks[cur][0][0], sub * 16, 64, 32, lane);
      sacc = wmma_bf16(kf.v, qf1.v, sacc);

      // binary key mask: lane holds 8 consecutive k rows starting at kb
      const int kb = kt + sub * 16 + half * 8;
      float4 mA = *(const float4*)(mrow + kb);
      float4 mB = *(const float4*)(mrow + kb + 4);
      sacc[0] *= mA.x; sacc[1] *= mA.y; sacc[2] *= mA.z; sacc[3] *= mA.w;
      sacc[4] *= mB.x; sacc[5] *= mB.y; sacc[6] *= mB.z; sacc[7] *= mB.w;

      // C-layout lane holds (q=m16 fixed, 8 consecutive k) -> row-major P
      *(uint4*)&Pl[wave][m16][sub * 16 + half * 8] = pack_bf16x8(sacc);
    }

    // ---- reload P as A-fragment (q rows x 32 k) ----
    FragA pf;
    load_fragA(pf, &Pl[wave][0][0], 0, 40, 0, lane);

    // ---- ctx^T += V^T_tile @ P^T  (4 hd subtiles) ----
#pragma unroll
    for (int t = 0; t < 4; ++t) {
      FragA vf;
      load_fragA(vf, &Vs[cur][0][0], t * 16, 32, 0, lane);
      ctx[t] = wmma_bf16(vf.v, pf.v, ctx[t]);
    }

    finish_tiles();
  }

  // ---- store ctx^T: lane has fixed q, 8 consecutive hd per tile ----
  const int qg = q0 + m16;
  float* orow = out + ((size_t)b * 2048 + qg) * 1024 + h * 64;
#pragma unroll
  for (int t = 0; t < 4; ++t) {
    const int hd = t * 16 + half * 8;
    float4 lo = { ctx[t][0], ctx[t][1], ctx[t][2], ctx[t][3] };
    float4 hi = { ctx[t][4], ctx[t][5], ctx[t][6], ctx[t][7] };
    *(float4*)(orow + hd)     = lo;
    *(float4*)(orow + hd + 4) = hi;
  }
}

// ---------------------------------------------------------------------------
extern "C" void kernel_launch(void* const* d_in, const int* in_sizes, int n_in,
                              void* d_out, int out_size, void* d_ws, size_t ws_size,
                              hipStream_t stream) {
  (void)in_sizes; (void)n_in; (void)out_size; (void)ws_size;
  const float* hs = (const float*)d_in[0];
  const float* am = (const float*)d_in[1];
  const float* Wq = (const float*)d_in[2];
  const float* bq = (const float*)d_in[3];
  const float* Wk = (const float*)d_in[4];
  const float* bk = (const float*)d_in[5];
  const float* Wv = (const float*)d_in[6];
  const float* bv = (const float*)d_in[7];
  float* out = (float*)d_out;

  constexpr size_t nX = (size_t)4096 * 1024;        // B*S x D
  constexpr size_t nW = (size_t)1024 * 1024;        // D x D
  constexpr size_t nH = (size_t)2 * 16 * 2048 * 64; // B*H*S*HD

  char* ws = (char*)d_ws;
  __bf16* Xbf = (__bf16*)ws;
  __bf16* Wqb = Xbf + nX;
  __bf16* Wkb = Wqb + nW;
  __bf16* Wvb = Wkb + nW;
  __bf16* Qbf = Wvb + nW;
  __bf16* Kbf = Qbf + nH;
  __bf16* Vtb = Kbf + nH;
  float*  mbin = (float*)(Vtb + nH);

  // Stage 0: converts
  cvt_f32_to_bf16<<<(unsigned)(nX / 4 / 256), 256, 0, stream>>>(hs, Xbf, (int)(nX / 4));
  cvt_f32_to_bf16<<<(unsigned)(nW / 4 / 256), 256, 0, stream>>>(Wq, Wqb, (int)(nW / 4));
  cvt_f32_to_bf16<<<(unsigned)(nW / 4 / 256), 256, 0, stream>>>(Wk, Wkb, (int)(nW / 4));
  cvt_f32_to_bf16<<<(unsigned)(nW / 4 / 256), 256, 0, stream>>>(Wv, Wvb, (int)(nW / 4));
  binarize_mask<<<16, 256, 0, stream>>>(am, mbin, 4096);

  // Stage 1: projections. mode0 grid: N=4096 -> 32, M=1024 -> 16.
  qkv_gemm<<<dim3(32, 16), 256, 0, stream>>>(Wqb, Xbf, bq, Qbf, 0);
  qkv_gemm<<<dim3(32, 16), 256, 0, stream>>>(Wkb, Xbf, bk, Kbf, 0);
  // mode1 grid: N=1024 -> 8, M=4096 -> 64.
  qkv_gemm<<<dim3(8, 64), 256, 0, stream>>>(Xbf, Wvb, bv, Vtb, 1);

  // Stage 2: attention. (B*H)=32 x (S/64)=32 -> 1024 WGs of 4 waves.
  attn_kernel<<<1024, 128, 0, stream>>>(Qbf, Kbf, Vtb, mbin, out);
}
